// CrossAttention_69544110457184
// MI455X (gfx1250) — matrix operand order repaired and checked by reading, not compiled, not run on hardware
//
#include <hip/hip_runtime.h>
#include <cstdint>
#include <cstddef>

#define B_    4
#define TQ_   1024
#define TKV_  1024
#define PAST_ 2048
#define C_    512
#define H_    8
#define HD_   64
#define TTOT_ 3072
#define SCALE_ 0.125f   // 1/sqrt(64)

typedef __attribute__((ext_vector_type(16))) __bf16 bf16x16;
typedef __attribute__((ext_vector_type(8)))  float  f32x8;

struct alignas(16) U128 { unsigned int x[4]; };

union Frag32B {
    U128 q[2];
    unsigned short us[16];
    unsigned int   ui[8];
    bf16x16        v;
};

__device__ __forceinline__ unsigned short f2bf(float f) {
    unsigned int x = __float_as_uint(f);
    x += 0x7FFFu + ((x >> 16) & 1u);          // round-to-nearest-even
    return (unsigned short)(x >> 16);
}
__device__ __forceinline__ unsigned int pack2bf(float a, float b) {
    return (unsigned int)f2bf(a) | ((unsigned int)f2bf(b) << 16);
}

// ---------------------------------------------------------------------------
// past_k (B,H,PAST,HD) f32 -> k_all bf16 [b][h][k][d]
// past_v (B,H,PAST,HD) f32 -> v_allT bf16 [b][h][d][k]   (transposed)
// ---------------------------------------------------------------------------
__global__ void cast_past_kernel(const float* __restrict__ pk,
                                 const float* __restrict__ pv,
                                 unsigned short* __restrict__ kall,
                                 unsigned short* __restrict__ vallT) {
    const int total = B_ * H_ * PAST_ * HD_;
    for (int idx = blockIdx.x * blockDim.x + threadIdx.x; idx < total;
         idx += gridDim.x * blockDim.x) {
        int d  = idx & (HD_ - 1);
        int k  = (idx >> 6) & (PAST_ - 1);
        int bh = idx >> 17;                    // / (PAST_*HD_)
        kall[((size_t)bh * TTOT_ + k) * HD_ + d]  = f2bf(pk[idx]);
        vallT[((size_t)bh * HD_ + d) * TTOT_ + k] = f2bf(pv[idx]);
    }
}

// ---------------------------------------------------------------------------
// Generic projection GEMM: out[M=4096, N=512] = A[M,512] * W[512,512] + bias
// MODE 0: -> Qb  bf16  [b][h][t][d]
// MODE 1: -> Kc  bf16  [b][h][PAST+t][d]
// MODE 2: -> Vc  bf16  [b][h][d][PAST+t]
// MODE 3: -> out f32   [m][n]
// ---------------------------------------------------------------------------
template <int MODE>
__global__ __launch_bounds__(256) void proj_gemm_kernel(
    const float* __restrict__ A, const float* __restrict__ W,
    const float* __restrict__ bias,
    unsigned short* __restrict__ obf, float* __restrict__ of32) {

    __shared__ unsigned short As[128 * 40];    // A tile   [m][k], bf16
    __shared__ unsigned short Bs[128 * 40];    // W tile^T [n][k], bf16

    const int tid  = threadIdx.x;
    const int lane = tid & 31, wid = tid >> 5;
    const int hh = lane >> 4, lr = lane & 15;
    const int wm = (wid >> 2) * 64;            // wave row block (0 / 64)
    const int wn = (wid & 3) * 32;             // wave col block (0..96)
    const int m0 = blockIdx.x * 128;
    const int n0 = blockIdx.y * 128;

    f32x8 acc[4][2];
    #pragma unroll
    for (int i = 0; i < 4; ++i)
        #pragma unroll
        for (int j = 0; j < 2; ++j)
            #pragma unroll
            for (int r = 0; r < 8; ++r) acc[i][j][r] = 0.f;

    const int arow = tid >> 1, acs = (tid & 1) * 16;
    const int bcol = tid & 127, bks = (tid >> 7) * 16;

    for (int kk = 0; kk < C_; kk += 32) {
        {   // stage A 128x32 (f32 -> bf16)
            const float* src = A + (size_t)(m0 + arow) * C_ + kk + acs;
            unsigned int* dst = (unsigned int*)&As[arow * 40 + acs];
            #pragma unroll
            for (int j = 0; j < 8; ++j)
                dst[j] = pack2bf(src[2 * j], src[2 * j + 1]);
        }
        {   // stage W^T 32x128 -> [n][k]
            const float* src = W + (size_t)(kk + bks) * C_ + n0 + bcol;
            unsigned int* dst = (unsigned int*)&Bs[bcol * 40 + bks];
            #pragma unroll
            for (int j = 0; j < 8; ++j)
                dst[j] = pack2bf(src[(2 * j) * C_], src[(2 * j + 1) * C_]);
        }
        __syncthreads();

        Frag32B af[4], bfr[2];
        #pragma unroll
        for (int mi = 0; mi < 4; ++mi) {   // A frag: lane=row, K {hh*8+j, 16+hh*8+j}
            const unsigned short* base = &As[(wm + mi * 16 + lr) * 40];
            af[mi].q[0] = *reinterpret_cast<const U128*>(base + hh * 8);
            af[mi].q[1] = *reinterpret_cast<const U128*>(base + 16 + hh * 8);
        }
        #pragma unroll
        for (int ni = 0; ni < 2; ++ni) {   // B frag: lane=col, K {hh*8+j, 16+hh*8+j}
            const unsigned short* base = &Bs[(wn + ni * 16 + lr) * 40];
            bfr[ni].q[0] = *reinterpret_cast<const U128*>(base + hh * 8);
            bfr[ni].q[1] = *reinterpret_cast<const U128*>(base + 16 + hh * 8);
        }
        #pragma unroll
        for (int mi = 0; mi < 4; ++mi)
            #pragma unroll
            for (int ni = 0; ni < 2; ++ni)
                acc[mi][ni] = __builtin_amdgcn_wmma_f32_16x16x32_bf16(
                    false, af[mi].v, false, bfr[ni].v, (short)0, acc[mi][ni],
                    false, false);
        __syncthreads();
    }

    // epilogue: C layout -> lane holds col n=lr, VGPR r -> row r + 8*hh
    #pragma unroll
    for (int mi = 0; mi < 4; ++mi) {
        #pragma unroll
        for (int ni = 0; ni < 2; ++ni) {
            const int n = n0 + wn + ni * 16 + lr;
            const float bv = bias[n];
            #pragma unroll
            for (int r = 0; r < 8; ++r) {
                const int m = m0 + wm + mi * 16 + r + 8 * hh;
                const float val = acc[mi][ni][r] + bv;
                if (MODE == 3) {
                    of32[(size_t)m * C_ + n] = val;
                } else {
                    const int b = m >> 10, t = m & 1023;
                    const int hd = n >> 6, d = n & 63;
                    const int bh = b * H_ + hd;
                    if (MODE == 0)
                        obf[((size_t)bh * TQ_ + t) * HD_ + d] = f2bf(val);
                    else if (MODE == 1)
                        obf[((size_t)bh * TTOT_ + PAST_ + t) * HD_ + d] = f2bf(val);
                    else
                        obf[((size_t)bh * HD_ + d) * TTOT_ + PAST_ + t] = f2bf(val);
                }
            }
        }
    }
}

// ---------------------------------------------------------------------------
// Flash attention: 4 waves/block, each wave owns a 16-row Q tile.
// y (f32, [b*TQ][C]) written ready for the output projection GEMM.
// ---------------------------------------------------------------------------
__global__ __launch_bounds__(128) void attn_kernel(
    const unsigned short* __restrict__ qb,
    const unsigned short* __restrict__ kall,
    const unsigned short* __restrict__ vallT,
    const float* __restrict__ mask,
    const int* __restrict__ vcl,
    float* __restrict__ yout) {

    __shared__ unsigned short Ps[4][16 * 40];  // per-wave P transpose tile

    const int wid = threadIdx.x >> 5, lane = threadIdx.x & 31;
    const int hh = lane >> 4, lr = lane & 15;
    const int q64 = blockIdx.x & 15;           // TQ/64 = 16
    const int bh  = blockIdx.x >> 4;           // 0..31
    const int b = bh >> 3, h = bh & 7;
    const int q0 = q64 * 64 + wid * 16;

    const unsigned short* qbase = qb   + (size_t)bh * TQ_ * HD_;
    const unsigned short* kbase = kall + (size_t)bh * TTOT_ * HD_;
    const unsigned short* vbase = vallT + (size_t)bh * HD_ * TTOT_;
    const int inv = PAST_ - vcl[b];
    unsigned short* myP = Ps[wid];

    // Q fragments (16 x 64 bf16 = two 16x32 A fragments), loaded once
    Frag32B qf0, qf1;
    {
        const unsigned short* qrow = qbase + (size_t)(q0 + lr) * HD_;
        qf0.q[0] = *reinterpret_cast<const U128*>(qrow + hh * 8);
        qf0.q[1] = *reinterpret_cast<const U128*>(qrow + 16 + hh * 8);
        qf1.q[0] = *reinterpret_cast<const U128*>(qrow + 32 + hh * 8);
        qf1.q[1] = *reinterpret_cast<const U128*>(qrow + 48 + hh * 8);
    }

    float mrow[8], lsum[8];
    f32x8 acc[4];
    #pragma unroll
    for (int r = 0; r < 8; ++r) { mrow[r] = -3.0e38f; lsum[r] = 0.f; }
    #pragma unroll
    for (int nt = 0; nt < 4; ++nt)
        #pragma unroll
        for (int r = 0; r < 8; ++r) acc[nt][r] = 0.f;

    for (int t = 0; t < TTOT_; t += 32) {
        // ---- S = Q * K^T  (two 16x16 tiles) ----
        f32x8 s[2];
        #pragma unroll
        for (int ct = 0; ct < 2; ++ct) {
            Frag32B kf0, kf1;  // B frags: lane=key col, K-dim = d
            const unsigned short* krow =
                kbase + (size_t)(t + ct * 16 + lr) * HD_;
            kf0.q[0] = *reinterpret_cast<const U128*>(krow + hh * 8);
            kf0.q[1] = *reinterpret_cast<const U128*>(krow + 16 + hh * 8);
            kf1.q[0] = *reinterpret_cast<const U128*>(krow + 32 + hh * 8);
            kf1.q[1] = *reinterpret_cast<const U128*>(krow + 48 + hh * 8);
            f32x8 z;
            #pragma unroll
            for (int r = 0; r < 8; ++r) z[r] = 0.f;
            z = __builtin_amdgcn_wmma_f32_16x16x32_bf16(
                false, qf0.v, false, kf0.v, (short)0, z, false, false);
            z = __builtin_amdgcn_wmma_f32_16x16x32_bf16(
                false, qf1.v, false, kf1.v, (short)0, z, false, false);
            s[ct] = z;
        }

        // ---- scale + mask + online softmax ----
        float p0v[8], p1v[8], alpha[8];
        #pragma unroll
        for (int r = 0; r < 8; ++r) {
            const int qg = q0 + r + 8 * hh;
            const int k0g = t + lr, k1g = t + 16 + lr;
            float v0 = s[0][r] * SCALE_ + mask[(size_t)qg * TTOT_ + k0g];
            float v1 = s[1][r] * SCALE_ + mask[(size_t)qg * TTOT_ + k1g];
            if (k0g < inv) v0 = -3.0e38f;
            if (k1g < inv) v1 = -3.0e38f;
            float mx = fmaxf(v0, v1);
            mx = fmaxf(mx, __shfl_xor(mx, 1, 32));
            mx = fmaxf(mx, __shfl_xor(mx, 2, 32));
            mx = fmaxf(mx, __shfl_xor(mx, 4, 32));
            mx = fmaxf(mx, __shfl_xor(mx, 8, 32));
            const float mnew = fmaxf(mrow[r], mx);
            const float a = __expf(mrow[r] - mnew);
            mrow[r] = mnew;
            alpha[r] = a;
            const float p0 = __expf(v0 - mnew);
            const float p1 = __expf(v1 - mnew);
            lsum[r] = lsum[r] * a + p0 + p1;   // lane-partial row sum
            p0v[r] = p0; p1v[r] = p1;
        }
        #pragma unroll
        for (int nt = 0; nt < 4; ++nt)
            #pragma unroll
            for (int r = 0; r < 8; ++r) acc[nt][r] *= alpha[r];

        // ---- transpose P (C layout -> A layout) through LDS ----
        #pragma unroll
        for (int r = 0; r < 8; ++r) {
            myP[(r + 8 * hh) * 40 + lr]      = f2bf(p0v[r]);
            myP[(r + 8 * hh) * 40 + 16 + lr] = f2bf(p1v[r]);
        }
        asm volatile("s_wait_dscnt 0" ::: "memory");
        Frag32B pf;
        {
            const unsigned short* prow = &myP[lr * 40];
            pf.q[0] = *reinterpret_cast<const U128*>(prow + hh * 8);
            pf.q[1] = *reinterpret_cast<const U128*>(prow + 16 + hh * 8);
        }

        // ---- y += P * V  (V stored transposed: contiguous in k) ----
        #pragma unroll
        for (int nt = 0; nt < 4; ++nt) {
            Frag32B vf;  // B frag: lane=col d, K-dim = key
            const unsigned short* vrow =
                vbase + (size_t)(nt * 16 + lr) * TTOT_ + t;
            vf.q[0] = *reinterpret_cast<const U128*>(vrow + hh * 8);
            vf.q[1] = *reinterpret_cast<const U128*>(vrow + 16 + hh * 8);
            acc[nt] = __builtin_amdgcn_wmma_f32_16x16x32_bf16(
                false, pf.v, false, vf.v, (short)0, acc[nt], false, false);
        }
    }

    // ---- normalize + store y (f32, [b*TQ][C]) ----
    float invl[8];
    #pragma unroll
    for (int r = 0; r < 8; ++r) {
        float ls = lsum[r];
        ls += __shfl_xor(ls, 1, 32);
        ls += __shfl_xor(ls, 2, 32);
        ls += __shfl_xor(ls, 4, 32);
        ls += __shfl_xor(ls, 8, 32);
        invl[r] = 1.0f / ls;
    }
    #pragma unroll
    for (int nt = 0; nt < 4; ++nt)
        #pragma unroll
        for (int r = 0; r < 8; ++r) {
            const size_t row = (size_t)b * TQ_ + q0 + r + 8 * hh;
            const int    col = h * HD_ + nt * 16 + lr;
            yout[row * C_ + col] = acc[nt][r] * invl[r];
        }
}

// ---------------------------------------------------------------------------
extern "C" void kernel_launch(void* const* d_in, const int* in_sizes, int n_in,
                              void* d_out, int out_size, void* d_ws, size_t ws_size,
                              hipStream_t stream) {
    const float* query = (const float*)d_in[0];
    const float* kvin  = (const float*)d_in[1];
    const float* pk    = (const float*)d_in[2];
    const float* pv    = (const float*)d_in[3];
    const float* mask  = (const float*)d_in[4];
    const int*   vcl   = (const int*)d_in[5];
    const float* Wq = (const float*)d_in[6];  const float* bq = (const float*)d_in[7];
    const float* Wk = (const float*)d_in[8];  const float* bk = (const float*)d_in[9];
    const float* Wv = (const float*)d_in[10]; const float* bv = (const float*)d_in[11];
    const float* Wp = (const float*)d_in[12]; const float* bp = (const float*)d_in[13];
    float* out = (float*)d_out;

    char* ws = (char*)d_ws;
    unsigned short* ws_q = (unsigned short*)ws;                        //  4 MB bf16 Q
    unsigned short* ws_k = (unsigned short*)(ws + (4u << 20));         // 12 MB bf16 K cache
    unsigned short* ws_v = (unsigned short*)(ws + (4u << 20) + 12582912u);
    float*          ws_y = (float*)(ws + (4u << 20) + 2u * 12582912u); //  8 MB f32 y

    cast_past_kernel<<<4096, 256, 0, stream>>>(pk, pv, ws_k, ws_v);

    dim3 g(32, 4);  // M=4096/128, N=512/128
    proj_gemm_kernel<0><<<g, 256, 0, stream>>>(query, Wq, bq, ws_q, nullptr);
    proj_gemm_kernel<1><<<g, 256, 0, stream>>>(kvin,  Wk, bk, ws_k, nullptr);
    proj_gemm_kernel<2><<<g, 256, 0, stream>>>(kvin,  Wv, bv, ws_v, nullptr);

    attn_kernel<<<512, 128, 0, stream>>>(ws_q, ws_k, ws_v, mask, vcl, ws_y);

    proj_gemm_kernel<3><<<g, 256, 0, stream>>>(ws_y, Wp, bp, nullptr, out);
}
